// ECGMambaNet_79714593014157
// MI455X (gfx1250) — matrix operand order repaired and checked by reading, not compiled, not run on hardware
//
#include <hip/hip_runtime.h>
#include <math.h>

// ---------------------------------------------------------------------------
// ECG Mamba network forward for MI455X (gfx1250, wave32, WMMA).
// Dense layers: v_wmma_f32_16x16x32_bf16, ping-pong software-pipelined K loop
// (no cross-buffer copies -> loads coalesce directly into WMMA source tuples),
// 1xNT strip per wave for A-fragment register reuse.
// ---------------------------------------------------------------------------

typedef __bf16 bf16;
typedef __attribute__((ext_vector_type(8)))  bf16  v8bf;
typedef __attribute__((ext_vector_type(16))) bf16  v16bf;
typedef __attribute__((ext_vector_type(8)))  float v8f;

#define NBATCH   16
#define LSEQ     2048
#define BL       (NBATCH * LSEQ)   // 32768 positions
#define DMODEL   256
#define DINNER   512
#define DSTATE   16
#define DTRANK   16

// A fragment: two 16B chunks at +0 and +16 elements (K {0..7,16..23} / {8..15,24..31})
__device__ __forceinline__ v16bf ld_fragA(const bf16* p) {
    v8bf lo = *(const v8bf*)p;
    v8bf hi = *(const v8bf*)(p + 16);
    return __builtin_shufflevector(lo, hi, 0,1,2,3,4,5,6,7,8,9,10,11,12,13,14,15);
}
// B fragment: two 16B chunks at +0 and +16 elements (same K order as A)
__device__ __forceinline__ v16bf ld_fragB(const bf16* p) {
    v8bf lo = *(const v8bf*)p;
    v8bf hi = *(const v8bf*)(p + 16);
    return __builtin_shufflevector(lo, hi, 0,1,2,3,4,5,6,7,8,9,10,11,12,13,14,15);
}

__device__ __forceinline__ v8f wmma_bf16(v16bf a, v16bf b, v8f c) {
    return __builtin_amdgcn_wmma_f32_16x16x32_bf16(
        /*neg_a=*/false, a, /*neg_b=*/false, b,
        /*c_mod=*/(short)0, c, /*reuse_a=*/false, /*reuse_b=*/false);
}

// ========================== WMMA GEMM ======================================
// C[M,N] = A[M,K] (bf16 row-major) * W[N,K]^T (bf16 row-major) [+bias] [+beta*C]
// One wave -> one 16x(16*NT) strip. Block = 256 threads = 8 waves = 8 M-tiles.
// Requires M%128==0, N%(16*NT)==0, K%32==0 (all call sites satisfy this).
template <typename OutT, int NT>
__global__ __launch_bounds__(256)
void gemm_wmma_bf16(const bf16* __restrict__ A,
                    const bf16* __restrict__ W,
                    const float* __restrict__ bias,
                    OutT* __restrict__ C,
                    int M, int N, int K, int ldc, int coff, int beta)
{
    const int wave = threadIdx.x >> 5;
    const int lane = threadIdx.x & 31;
    const int mt   = blockIdx.x * 8 + wave;
    const int ntg  = blockIdx.y * NT;          // first N tile of this strip

    const int l15  = lane & 15;
    const int hi   = lane >> 4;                // 0: lanes 0-15, 1: lanes 16-31

    const bf16* Aptr = A + (size_t)(mt * 16 + l15) * K + (hi << 3);
    const bf16* Wptr[NT];
#pragma unroll
    for (int j = 0; j < NT; ++j)
        Wptr[j] = W + (size_t)((ntg + j) * 16 + l15) * K + (hi << 3);

    v8f acc[NT];
#pragma unroll
    for (int j = 0; j < NT; ++j) acc[j] = (v8f){};

    // ---- prologue: chunk 0 into buffer set "a" ----
    v16bf Aa = ld_fragA(Aptr);
    v16bf Ba[NT];
#pragma unroll
    for (int j = 0; j < NT; ++j) Ba[j] = ld_fragB(Wptr[j]);

    // ---- ping-pong main loop: two 32-wide K chunks per iteration ----
    int k0 = 0;
    for (; k0 + 64 <= K; k0 += 64) {
        // chunk k0+32 -> buffer set "b" (in flight while "a" computes)
        v16bf Ab = ld_fragA(Aptr + k0 + 32);
        v16bf Bb[NT];
#pragma unroll
        for (int j = 0; j < NT; ++j) Bb[j] = ld_fragB(Wptr[j] + k0 + 32);
        if (k0 + 96 < K) {
            __builtin_prefetch((const void*)(Aptr + k0 + 96), 0, 1);
            __builtin_prefetch((const void*)(Wptr[0] + k0 + 96), 0, 1);
        }
#pragma unroll
        for (int j = 0; j < NT; ++j) acc[j] = wmma_bf16(Aa, Ba[j], acc[j]);
        // chunk k0+64 -> buffer set "a" (old "a" is dead: loads land in place)
        if (k0 + 64 < K) {
            Aa = ld_fragA(Aptr + k0 + 64);
#pragma unroll
            for (int j = 0; j < NT; ++j) Ba[j] = ld_fragB(Wptr[j] + k0 + 64);
        }
#pragma unroll
        for (int j = 0; j < NT; ++j) acc[j] = wmma_bf16(Ab, Bb[j], acc[j]);
    }
    if (k0 < K) {   // odd chunk count (e.g. K=32): last chunk sits in "a"
#pragma unroll
        for (int j = 0; j < NT; ++j) acc[j] = wmma_bf16(Aa, Ba[j], acc[j]);
    }

    // ---- epilogue: C layout: VGPR r -> row mt*16 + r + 8*hi, col = lane&15 ----
    const int rbase = mt * 16 + (hi << 3);
#pragma unroll
    for (int j = 0; j < NT; ++j) {
        const int ncol = (ntg + j) * 16 + l15;
        const float bval = bias ? bias[ncol] : 0.0f;
        const int col = coff + ncol;
#pragma unroll
        for (int r = 0; r < 8; ++r) {
            size_t idx = (size_t)(rbase + r) * ldc + col;
            float v = acc[j][r] + bval;
            if (beta) v += (float)C[idx];
            C[idx] = (OutT)v;
        }
    }
}

// ========================== elementwise / small kernels ====================
__device__ __forceinline__ float gelu_exact(float v) {
    return 0.5f * v * (1.0f + erff(v * 0.70710678118654752f));
}
__device__ __forceinline__ float silu_f(float v) {
    return v / (1.0f + __expf(-v));
}

__global__ void f32_to_bf16_k(const float* __restrict__ s, bf16* __restrict__ d, int n) {
    int i = blockIdx.x * blockDim.x + threadIdx.x;
    if (i < n) d[i] = (bf16)s[i];
}

// dt_proj_w (512x16) -> bf16 (512x32) zero-padded in K for WMMA alignment
__global__ void pad_dtw_k(const float* __restrict__ s, bf16* __restrict__ d) {
    int i = blockIdx.x * blockDim.x + threadIdx.x;       // 512*32
    if (i >= DINNER * 32) return;
    int nrow = i >> 5, k = i & 31;
    d[i] = (k < DTRANK) ? (bf16)s[nrow * DTRANK + k] : (bf16)0.0f;
}

// stem conv1 (12->64, kw=7, pad 3) + BN + gelu, output position-major bf16
__global__ void stem_conv1_k(const float* __restrict__ x, const float* __restrict__ w1,
                             const float* __restrict__ g, const float* __restrict__ b,
                             const float* __restrict__ m, const float* __restrict__ v,
                             bf16* __restrict__ out)
{
    int tid = blockIdx.x * blockDim.x + threadIdx.x;     // NBATCH*64*LSEQ
    if (tid >= NBATCH * 64 * LSEQ) return;
    int bb = tid / (64 * LSEQ);
    int rest = tid % (64 * LSEQ);
    int oc = rest / LSEQ, l = rest % LSEQ;
    float acc = 0.0f;
    for (int ic = 0; ic < 12; ++ic) {
        const float* xp = x + ((size_t)bb * 12 + ic) * LSEQ;
        const float* wp = w1 + ((size_t)oc * 12 + ic) * 7;
#pragma unroll
        for (int k = 0; k < 7; ++k) {
            int idx = l + k - 3;
            if (idx >= 0 && idx < LSEQ) acc += xp[idx] * wp[k];
        }
    }
    float vv = (acc - m[oc]) * rsqrtf(v[oc] + 1e-5f) * g[oc] + b[oc];
    out[((size_t)bb * LSEQ + l) * 64 + oc] = (bf16)gelu_exact(vv);
}

// BN2 + gelu, in place on f32 [BL x 256]
__global__ void bn2_gelu_k(float* __restrict__ h, const float* __restrict__ g,
                           const float* __restrict__ b, const float* __restrict__ m,
                           const float* __restrict__ v)
{
    int i = blockIdx.x * blockDim.x + threadIdx.x;
    if (i >= BL * DMODEL) return;
    int oc = i % DMODEL;
    float vv = (h[i] - m[oc]) * rsqrtf(v[oc] + 1e-5f) * g[oc] + b[oc];
    h[i] = gelu_exact(vv);
}

// layernorm over last dim (256), f32 in -> bf16 out, one thread per row
__global__ void layernorm_k(const float* __restrict__ x, const float* __restrict__ g,
                            const float* __restrict__ b, bf16* __restrict__ out, int rows)
{
    int r = blockIdx.x * blockDim.x + threadIdx.x;
    if (r >= rows) return;
    const float* xr = x + (size_t)r * DMODEL;
    float mu = 0.0f;
    for (int i = 0; i < DMODEL; ++i) mu += xr[i];
    mu *= (1.0f / DMODEL);
    float var = 0.0f;
    for (int i = 0; i < DMODEL; ++i) { float d = xr[i] - mu; var += d * d; }
    var *= (1.0f / DMODEL);
    float rs = rsqrtf(var + 1e-5f);
    bf16* o = out + (size_t)r * DMODEL;
    for (int i = 0; i < DMODEL; ++i) o[i] = (bf16)((xr[i] - mu) * rs * g[i] + b[i]);
}

// depthwise causal conv (kw=4) + silu; dir=+1 fwd, -1 bwd (flip-free).
// xi lives in columns 0..511 of xz (ld 1024).
__global__ void dwconv_silu_k(const bf16* __restrict__ xz, const float* __restrict__ cw,
                              const float* __restrict__ cb, bf16* __restrict__ out, int dir)
{
    int i = blockIdx.x * blockDim.x + threadIdx.x;       // BL*512
    if (i >= BL * DINNER) return;
    int row = i / DINNER, d = i % DINNER;
    int bb = row / LSEQ, l = row % LSEQ;
    float acc = cb[d];
#pragma unroll
    for (int k = 0; k < 4; ++k) {
        int idx = l + dir * (k - 3);
        if (idx >= 0 && idx < LSEQ)
            acc += (float)xz[((size_t)bb * LSEQ + idx) * (2 * DINNER) + d] * cw[d * 4 + k];
    }
    out[(size_t)row * DINNER + d] = (bf16)silu_f(acc);
}

// dt_low: first 16 cols of x_dbl (f32, ld 48) -> bf16 [BL x 32] zero-padded
__global__ void make_dtlow_k(const float* __restrict__ xdbl, bf16* __restrict__ out) {
    int i = blockIdx.x * blockDim.x + threadIdx.x;       // BL*32
    if (i >= BL * 32) return;
    int row = i >> 5, c = i & 31;
    out[i] = (c < DTRANK) ? (bf16)xdbl[(size_t)row * 48 + c] : (bf16)0.0f;
}

__global__ void softplus_k(bf16* __restrict__ p, int n) {
    int i = blockIdx.x * blockDim.x + threadIdx.x;
    if (i >= n) return;
    float v = (float)p[i];
    p[i] = (bf16)(v > 20.0f ? v : log1pf(__expf(v)));
}

// selective scan: one thread per (batch, channel); 16-wide state in registers.
__global__ void scan_k(const bf16* __restrict__ xc, const bf16* __restrict__ dtb,
                       const float* __restrict__ xdbl, const float* __restrict__ A_log,
                       const float* __restrict__ Dp, bf16* __restrict__ y, int dir)
{
    int tid = blockIdx.x * blockDim.x + threadIdx.x;
    if (tid >= NBATCH * DINNER) return;
    int bb = tid / DINNER, d = tid % DINNER;
    float a[DSTATE], h[DSTATE];
#pragma unroll
    for (int n = 0; n < DSTATE; ++n) { a[n] = -__expf(A_log[d * DSTATE + n]); h[n] = 0.0f; }
    const float Dd = Dp[d];
    for (int t = 0; t < LSEQ; ++t) {
        int l = (dir > 0) ? t : (LSEQ - 1 - t);
        size_t row = (size_t)bb * LSEQ + l;
        float dt = (float)dtb[row * DINNER + d];
        float xv = (float)xc[row * DINNER + d];
        const float* xr = xdbl + row * 48;
        float yv = 0.0f;
#pragma unroll
        for (int n = 0; n < DSTATE; ++n) {
            float Bn = xr[16 + n], Cn = xr[32 + n];
            h[n] = __expf(dt * a[n]) * h[n] + dt * xv * Bn;
            yv += h[n] * Cn;
        }
        y[row * DINNER + d] = (bf16)(yv + xv * Dd);
    }
}

// y *= silu(z); z is cols 512..1023 of xz (ld 1024)
__global__ void gate_k(bf16* __restrict__ y, const bf16* __restrict__ xz) {
    int i = blockIdx.x * blockDim.x + threadIdx.x;
    if (i >= BL * DINNER) return;
    int row = i / DINNER, d = i % DINNER;
    float z = (float)xz[(size_t)row * (2 * DINNER) + DINNER + d];
    y[i] = (bf16)((float)y[i] * silu_f(z));
}

// mean over L of layernormed rows (bf16 in) -> pooled f32 [16 x 256]
__global__ void pool_k(const bf16* __restrict__ hn, float* __restrict__ pooled) {
    int i = blockIdx.x * blockDim.x + threadIdx.x;       // 16*256
    if (i >= NBATCH * DMODEL) return;
    int bb = i / DMODEL, c = i % DMODEL;
    float s = 0.0f;
    for (int l = 0; l < LSEQ; ++l)
        s += (float)hn[((size_t)bb * LSEQ + l) * DMODEL + c];
    pooled[i] = s * (1.0f / LSEQ);
}

// classifier 16x9
__global__ void cls_k(const float* __restrict__ pooled, const float* __restrict__ w,
                      const float* __restrict__ b, float* __restrict__ out) {
    int i = blockIdx.x * blockDim.x + threadIdx.x;       // 144
    if (i >= NBATCH * 9) return;
    int bb = i / 9, j = i % 9;
    float s = b[j];
    for (int c = 0; c < DMODEL; ++c) s += pooled[bb * DMODEL + c] * w[j * DMODEL + c];
    out[i] = s;
}

// ========================== host orchestration =============================
template <typename OutT>
static void launch_gemm(const bf16* A, const bf16* W, const float* bias, OutT* C,
                        int M, int N, int K, int ldc, int coff, int beta, hipStream_t s)
{
    dim3 b(256);
    if (N % 64 == 0) {
        dim3 g(M / 128, N / 64);
        gemm_wmma_bf16<OutT, 4><<<g, b, 0, s>>>(A, W, bias, C, M, N, K, ldc, coff, beta);
    } else if (N % 48 == 0) {
        dim3 g(M / 128, N / 48);
        gemm_wmma_bf16<OutT, 3><<<g, b, 0, s>>>(A, W, bias, C, M, N, K, ldc, coff, beta);
    } else {
        dim3 g(M / 128, N / 16);
        gemm_wmma_bf16<OutT, 1><<<g, b, 0, s>>>(A, W, bias, C, M, N, K, ldc, coff, beta);
    }
}

struct MambaW {
    const float *A_log, *D, *conv_b, *conv_w, *dt_b, *dt_w, *in_w, *out_w, *x_w;
};

extern "C" void kernel_launch(void* const* d_in, const int* in_sizes, int n_in,
                              void* d_out, int out_size, void* d_ws, size_t ws_size,
                              hipStream_t stream)
{
    (void)in_sizes; (void)n_in; (void)out_size; (void)ws_size;
    const float* const* in = (const float* const*)d_in;

    // JAX pytree (sorted dict keys) flattening:
    // per block (22): bwd[9], fwd[9], norm_b, norm_g, proj_b, proj_w
    // mamba (9): A_log, D, conv_b, conv_w, dt_proj_b, dt_proj_w, in_proj_w, out_proj_w, x_proj_w
    auto getm = [&](int base) {
        MambaW m;
        m.A_log = in[base + 0]; m.D    = in[base + 1]; m.conv_b = in[base + 2];
        m.conv_w = in[base + 3]; m.dt_b = in[base + 4]; m.dt_w  = in[base + 5];
        m.in_w  = in[base + 6]; m.out_w = in[base + 7]; m.x_w   = in[base + 8];
        return m;
    };
    const float* head_cls_b = in[66];
    const float* head_cls_w = in[67];
    const float* head_nb    = in[68];
    const float* head_ng    = in[69];
    const float* bn1_b = in[70]; const float* bn1_g = in[71];
    const float* bn1_m = in[72]; const float* bn1_v = in[73];
    const float* bn2_b = in[74]; const float* bn2_g = in[75];
    const float* bn2_m = in[76]; const float* bn2_v = in[77];
    const float* w1 = in[78];    const float* w2 = in[79];
    const float* x  = in[80];
    float* out = (float*)d_out;

    // ---- workspace carve ----
    char* p = (char*)d_ws;
    auto carve = [&](size_t bytes) -> void* {
        void* r = (void*)p; p += (bytes + 255) & ~(size_t)255; return r;
    };
    float* h      = (float*)carve((size_t)BL * DMODEL * 4);       // residual stream
    bf16*  hn     = (bf16*) carve((size_t)BL * DMODEL * 2);
    bf16*  act1   = (bf16*) carve((size_t)BL * 64 * 2);
    bf16*  xz     = (bf16*) carve((size_t)BL * 2 * DINNER * 2);
    bf16*  xc     = (bf16*) carve((size_t)BL * DINNER * 2);
    float* xdbl   = (float*)carve((size_t)BL * 48 * 4);
    bf16*  dtlow  = (bf16*) carve((size_t)BL * 32 * 2);
    bf16*  dtb    = (bf16*) carve((size_t)BL * DINNER * 2);
    bf16*  yb     = (bf16*) carve((size_t)BL * DINNER * 2);
    bf16*  cat    = (bf16*) carve((size_t)BL * DINNER * 2);
    float* pooled = (float*)carve((size_t)NBATCH * DMODEL * 4);
    bf16*  w2b    = (bf16*) carve((size_t)DMODEL * 64 * 2);
    bf16*  inwb[3][2], *outwb[3][2], *xwb[3][2], *dtwb[3][2], *projwb[3];
    for (int blk = 0; blk < 3; ++blk) {
        for (int dir = 0; dir < 2; ++dir) {
            inwb[blk][dir]  = (bf16*)carve((size_t)2 * DINNER * DMODEL * 2);
            outwb[blk][dir] = (bf16*)carve((size_t)DMODEL * DINNER * 2);
            xwb[blk][dir]   = (bf16*)carve((size_t)(DTRANK + 2 * DSTATE) * DINNER * 2);
            dtwb[blk][dir]  = (bf16*)carve((size_t)DINNER * 32 * 2);
        }
        projwb[blk] = (bf16*)carve((size_t)DMODEL * DINNER * 2);
    }

    auto cvt = [&](const float* s, bf16* d, int n) {
        f32_to_bf16_k<<<(n + 255) / 256, 256, 0, stream>>>(s, d, n);
    };

    // ---- weight conversion (every call; deterministic, no caching) ----
    cvt(w2, w2b, DMODEL * 64);
    for (int blk = 0; blk < 3; ++blk) {
        int base = blk * 22;
        MambaW mw[2] = { getm(base + 9) /*fwd*/, getm(base + 0) /*bwd*/ };
        for (int dir = 0; dir < 2; ++dir) {
            cvt(mw[dir].in_w,  inwb[blk][dir],  2 * DINNER * DMODEL);
            cvt(mw[dir].out_w, outwb[blk][dir], DMODEL * DINNER);
            cvt(mw[dir].x_w,   xwb[blk][dir],   (DTRANK + 2 * DSTATE) * DINNER);
            pad_dtw_k<<<(DINNER * 32 + 255) / 256, 256, 0, stream>>>(mw[dir].dt_w, dtwb[blk][dir]);
        }
        cvt(in[base + 21], projwb[blk], DMODEL * DINNER);
    }

    // ---- stem ----
    {
        int n1 = NBATCH * 64 * LSEQ;
        stem_conv1_k<<<(n1 + 255) / 256, 256, 0, stream>>>(x, w1, bn1_g, bn1_b, bn1_m, bn1_v, act1);
        launch_gemm<float>(act1, w2b, nullptr, h, BL, DMODEL, 64, DMODEL, 0, 0, stream);
        bn2_gelu_k<<<(BL * DMODEL + 255) / 256, 256, 0, stream>>>(h, bn2_g, bn2_b, bn2_m, bn2_v);
    }

    // ---- mamba blocks ----
    for (int blk = 0; blk < 3; ++blk) {
        int base = blk * 22;
        const float* nrm_b = in[base + 18];
        const float* nrm_g = in[base + 19];
        const float* prj_b = in[base + 20];
        MambaW mw[2] = { getm(base + 9) /*fwd*/, getm(base + 0) /*bwd*/ };

        layernorm_k<<<(BL + 255) / 256, 256, 0, stream>>>(h, nrm_g, nrm_b, hn, BL);

        for (int di = 0; di < 2; ++di) {
            const int dir = (di == 0) ? +1 : -1;
            const MambaW& m = mw[di];
            // in_proj: [BL,256] x [1024,256]^T -> xz bf16 [BL,1024]
            launch_gemm<bf16>(hn, inwb[blk][di], nullptr, xz, BL, 2 * DINNER, DMODEL, 2 * DINNER, 0, 0, stream);
            // depthwise conv + silu (direction-aware, flip-free)
            dwconv_silu_k<<<(BL * DINNER + 255) / 256, 256, 0, stream>>>(xz, m.conv_w, m.conv_b, xc, dir);
            // x_proj: [BL,512] x [48,512]^T -> x_dbl f32 [BL,48]
            launch_gemm<float>(xc, xwb[blk][di], nullptr, xdbl, BL, 48, DINNER, 48, 0, 0, stream);
            // dt = softplus(dt_low @ dt_proj_w^T + b)
            make_dtlow_k<<<(BL * 32 + 255) / 256, 256, 0, stream>>>(xdbl, dtlow);
            launch_gemm<bf16>(dtlow, dtwb[blk][di], m.dt_b, dtb, BL, DINNER, 32, DINNER, 0, 0, stream);
            softplus_k<<<(BL * DINNER + 255) / 256, 256, 0, stream>>>(dtb, BL * DINNER);
            // selective scan (sequential in L, parallel over B*D_INNER)
            scan_k<<<(NBATCH * DINNER + 255) / 256, 256, 0, stream>>>(xc, dtb, xdbl, m.A_log, m.D, yb, dir);
            // gate with silu(z)
            gate_k<<<(BL * DINNER + 255) / 256, 256, 0, stream>>>(yb, xz);
            // out_proj -> concat buffer half (coff 0 = fwd, 256 = bwd)
            launch_gemm<bf16>(yb, outwb[blk][di], nullptr, cat, BL, DMODEL, DINNER, 2 * DMODEL, di * DMODEL, 0, stream);
        }
        // h += concat @ proj_w^T + proj_b   (beta = 1 residual accumulate)
        launch_gemm<float>(cat, projwb[blk], prj_b, h, BL, DMODEL, 2 * DMODEL, DMODEL, 0, 1, stream);
    }

    // ---- head ----
    layernorm_k<<<(BL + 255) / 256, 256, 0, stream>>>(h, head_ng, head_nb, hn, BL);
    pool_k<<<(NBATCH * DMODEL + 255) / 256, 256, 0, stream>>>(hn, pooled);
    cls_k<<<(NBATCH * 9 + 255) / 256, 256, 0, stream>>>(pooled, head_cls_w, head_cls_b, out);
}
